// GLU_55731495633054
// MI455X (gfx1250) — matrix operand order repaired and checked by reading, not compiled, not run on hardware
//
#include <hip/hip_runtime.h>
#include <hip/hip_bf16.h>

typedef __attribute__((ext_vector_type(16))) _Float16 v16h;
typedef __attribute__((ext_vector_type(8)))  _Float16 v8h;
typedef __attribute__((ext_vector_type(8)))  float    v8f;

#define DIM_IN   2048
#define DIM_OUT  8192
#define N_TOK    8192
#define NHALF    4096
#define W_ELEMS  (DIM_IN * DIM_OUT)   // 16,777,216

// ---- workspace layout (bytes) ----
#define WS_WSUM   0          // float   : sum |weight|
#define WS_GBITS  4          // uint    : bits of max|normed| (>=0 so uint-max works)
#define WS_PART   16         // 256 floats : partial sums
#define WS_MEAN   4096       // 8192 floats
#define WS_RSTD   (WS_MEAN + N_TOK * 4)           // 8192 floats
#define WS_QA     (WS_RSTD + N_TOK * 4)           // 8192x2048 f16
#define WS_WB     (WS_QA + (size_t)W_ELEMS * 2)   // 8192x2048 f16

// LDS B-tile: 128 rows (64 h-cols + 64 gate-cols) x 32 f16 (64B),
// padded to 80B/row so the 16 row-lanes start on distinct banks.
#define BROW_BYTES 80
#define BBUF_BYTES (128 * BROW_BYTES)   // 10240 B per buffer

// -------------------------------------------------------------------------
// 1a) per-block partial sums of |weight| (fixed work assignment -> determ.)
__global__ void wsum_partial_kernel(const float* __restrict__ w,
                                    float* __restrict__ partials) {
    __shared__ float red[256];
    int t = threadIdx.x;
    float s = 0.0f;
    int base = (blockIdx.x * 256 + t) * 256;
    const float4* w4 = (const float4*)(w + base);
    #pragma unroll 4
    for (int i = 0; i < 64; ++i) {
        float4 v = w4[i];
        s += fabsf(v.x) + fabsf(v.y) + fabsf(v.z) + fabsf(v.w);
    }
    red[t] = s; __syncthreads();
    for (int off = 128; off > 0; off >>= 1) {
        if (t < off) red[t] += red[t + off];
        __syncthreads();
    }
    if (t == 0) partials[blockIdx.x] = red[0];
}

// 1b) final fixed-order reduction of the 256 partials
__global__ void wsum_final_kernel(const float* __restrict__ partials,
                                  float* __restrict__ wsum) {
    __shared__ float red[256];
    int t = threadIdx.x;
    red[t] = partials[t]; __syncthreads();
    for (int off = 128; off > 0; off >>= 1) {
        if (t < off) red[t] += red[t + off];
        __syncthreads();
    }
    if (t == 0) *wsum = red[0];
}

// -------------------------------------------------------------------------
// 2) per-row layernorm stats + global max|normed| (atomic max: order-indep.)
__global__ void ln_stats_kernel(const float* __restrict__ x,
                                float* __restrict__ mean,
                                float* __restrict__ rstd,
                                unsigned int* __restrict__ gbits) {
    __shared__ float red[256];
    int row = blockIdx.x;
    int t   = threadIdx.x;
    const float* xr = x + (size_t)row * DIM_IN;
    float4 v0 = ((const float4*)xr)[t * 2 + 0];
    float4 v1 = ((const float4*)xr)[t * 2 + 1];
    float vals[8] = {v0.x, v0.y, v0.z, v0.w, v1.x, v1.y, v1.z, v1.w};

    float s = 0.0f, ss = 0.0f;
    #pragma unroll
    for (int i = 0; i < 8; ++i) { s += vals[i]; ss += vals[i] * vals[i]; }

    red[t] = s; __syncthreads();
    for (int off = 128; off > 0; off >>= 1) {
        if (t < off) red[t] += red[t + off];
        __syncthreads();
    }
    float mu = red[0] * (1.0f / DIM_IN);
    __syncthreads();
    red[t] = ss; __syncthreads();
    for (int off = 128; off > 0; off >>= 1) {
        if (t < off) red[t] += red[t + off];
        __syncthreads();
    }
    float var = red[0] * (1.0f / DIM_IN) - mu * mu;
    var = fmaxf(var, 0.0f);
    float rs = rsqrtf(var + 1e-5f);
    if (t == 0) { mean[row] = mu; rstd[row] = rs; }

    float mx = 0.0f;
    #pragma unroll
    for (int i = 0; i < 8; ++i) mx = fmaxf(mx, fabsf((vals[i] - mu) * rs));
    __syncthreads();
    red[t] = mx; __syncthreads();
    for (int off = 128; off > 0; off >>= 1) {
        if (t < off) red[t] = fmaxf(red[t], red[t + off]);
        __syncthreads();
    }
    if (t == 0) atomicMax(gbits, __float_as_uint(red[0]));
}

// -------------------------------------------------------------------------
// 3) quantize activations: clip(normed*128/gamma, +-(128-1e-5)) -> f16
__global__ void quant_act_kernel(const float* __restrict__ x,
                                 const float* __restrict__ mean,
                                 const float* __restrict__ rstd,
                                 const unsigned int* __restrict__ gbits,
                                 _Float16* __restrict__ qa) {
    int row = blockIdx.x;
    int t   = threadIdx.x;
    float gamma = __uint_as_float(*gbits);
    float scl   = 128.0f / gamma;
    float mu = mean[row], rs = rstd[row];
    const float* xr = x + (size_t)row * DIM_IN;
    float4 v0 = ((const float4*)xr)[t * 2 + 0];
    float4 v1 = ((const float4*)xr)[t * 2 + 1];
    float vals[8] = {v0.x, v0.y, v0.z, v0.w, v1.x, v1.y, v1.z, v1.w};
    v8h q;
    #pragma unroll
    for (int i = 0; i < 8; ++i) {
        float qv = (vals[i] - mu) * rs * scl;
        qv = fminf(fmaxf(qv, -128.0f + 1e-5f), 128.0f - 1e-5f);
        q[i] = (_Float16)qv;
    }
    *(v8h*)(qa + (size_t)row * DIM_IN + t * 8) = q;
}

// -------------------------------------------------------------------------
// 4) ternarize weights: clip(rne(w/(w_gamma+1e-7)), -1, 1) -> f16
__global__ void tern_w_kernel(const float* __restrict__ w,
                              const float* __restrict__ wsum,
                              _Float16* __restrict__ wb) {
    int t = threadIdx.x;
    float wg   = (*wsum) * (1.0f / (float)W_ELEMS);
    float inv  = 1.0f / (wg + 1e-7f);
    int base = (blockIdx.x * 256 + t) * 8;
    float4 v0 = ((const float4*)(w + base))[0];
    float4 v1 = ((const float4*)(w + base))[1];
    float vals[8] = {v0.x, v0.y, v0.z, v0.w, v1.x, v1.y, v1.z, v1.w};
    v8h q;
    #pragma unroll
    for (int i = 0; i < 8; ++i) {
        float r = rintf(vals[i] * inv);           // round-to-nearest-even
        r = fminf(fmaxf(r, -1.0f), 1.0f);
        q[i] = (_Float16)r;
    }
    *(v8h*)(wb + base) = q;
}

// -------------------------------------------------------------------------
// CDNA5 async copy: 16 B global -> LDS per lane, tracked by ASYNCcnt.
__device__ __forceinline__ void async_copy_b128(unsigned lds_off,
                                                const _Float16* gptr) {
    asm volatile("global_load_async_to_lds_b128 %0, %1, off"
                 :: "v"(lds_off), "v"(gptr)
                 : "memory");
}
__device__ __forceinline__ void wait_asynccnt0() {
    asm volatile("s_wait_asynccnt 0x0" ::: "memory");
}

__device__ __forceinline__ v16h cat8(v8h a, v8h b) {
    return __builtin_shufflevector(a, b, 0,1,2,3,4,5,6,7,8,9,10,11,12,13,14,15);
}
__device__ __forceinline__ v16h ldfrag(const char* p) {
    return cat8(((const v8h*)p)[0], ((const v8h*)p)[1]);
}
__device__ __forceinline__ v16h loadA(const _Float16* ap, int hi) {
    return cat8(*(const v8h*)(ap + hi * 16),
                *(const v8h*)(ap + hi * 16 + 8));
}
#define WMMA_F16(a, b, c) __builtin_amdgcn_wmma_f32_16x16x32_f16( \
        false, (a), false, (b), (short)0, (c), false, false)

// One K=32 step: 8 WMMAs against the LDS B-tile at 'bp', loads pipelined.
__device__ __forceinline__ void kstep(v8f (&acc)[8], v16h a, const char* bp) {
    v16h b0 = ldfrag(bp + 0 * 16 * BROW_BYTES);   // h0
    v16h b1 = ldfrag(bp + 4 * 16 * BROW_BYTES);   // g0
    v16h b2 = ldfrag(bp + 1 * 16 * BROW_BYTES);   // h1
    acc[0] = WMMA_F16(a, b0, acc[0]);
    v16h b3 = ldfrag(bp + 5 * 16 * BROW_BYTES);   // g1
    acc[4] = WMMA_F16(a, b1, acc[4]);
    v16h b4 = ldfrag(bp + 2 * 16 * BROW_BYTES);   // h2
    acc[1] = WMMA_F16(a, b2, acc[1]);
    v16h b5 = ldfrag(bp + 6 * 16 * BROW_BYTES);   // g2
    acc[5] = WMMA_F16(a, b3, acc[5]);
    v16h b6 = ldfrag(bp + 3 * 16 * BROW_BYTES);   // h3
    acc[2] = WMMA_F16(a, b4, acc[2]);
    v16h b7 = ldfrag(bp + 7 * 16 * BROW_BYTES);   // g3
    acc[6] = WMMA_F16(a, b5, acc[6]);
    acc[3] = WMMA_F16(a, b6, acc[3]);
    acc[7] = WMMA_F16(a, b7, acc[7]);
}

// -------------------------------------------------------------------------
// 5) fused WMMA GEMM + scale + GLU (h * silu(gate))
//    block = 256 threads = 8 waves; block tile = 128 rows x (64 h + 64 gate)
//    B tile staged in LDS via async global->LDS, double buffered.
//    A fragment double-buffered in registers (a0/a1), K-loop unrolled x2.
__global__ void __launch_bounds__(256)
gemm_glu_kernel(const _Float16* __restrict__ qa,
                const _Float16* __restrict__ wb,
                const float* __restrict__ wsum,
                const unsigned int* __restrict__ gbits,
                float* __restrict__ out) {
    __shared__ __align__(16) char Bs[2 * BBUF_BYTES];

    const int K = DIM_IN;
    int t    = threadIdx.x;
    int lane = t & 31;
    int wave = t >> 5;
    int lo = lane & 15;
    int hi = lane >> 4;

    int m0 = blockIdx.y * 128 + wave * 16;
    int n0 = blockIdx.x * 64;

    // --- per-thread async-copy assignment: 2 x 16B transfers per K-step ---
    unsigned ldsbase = (unsigned)(size_t)&Bs[0];
    int j0 = t >> 2,         q0 = t & 3;
    int j1 = (t + 256) >> 2, q1 = t & 3;
    int col0 = (j0 < 64) ? (n0 + j0) : (n0 + NHALF + (j0 - 64));
    int col1 = (j1 < 64) ? (n0 + j1) : (n0 + NHALF + (j1 - 64));
    const _Float16* gp0 = wb + (size_t)col0 * K + q0 * 8;
    const _Float16* gp1 = wb + (size_t)col1 * K + q1 * 8;
    unsigned lp0 = ldsbase + (unsigned)(j0 * BROW_BYTES + q0 * 16);
    unsigned lp1 = ldsbase + (unsigned)(j1 * BROW_BYTES + q1 * 16);

    const _Float16* ap = qa + (size_t)(m0 + lo) * K;
    const char* bfrag = Bs + lo * BROW_BYTES + hi * 32;

    v8f acc[8];
    #pragma unroll
    for (int i = 0; i < 8; ++i) acc[i] = (v8f){};

    // prologue: stage K-chunk 0 into buffer 0; prefetch A for K-step 0
    async_copy_b128(lp0, gp0);
    async_copy_b128(lp1, gp1);
    v16h a0 = loadA(ap, hi);
    v16h a1;

    const int NK = K / 32;   // 64 K-steps, processed in pairs
    #pragma unroll 1
    for (int it = 0; it < NK; it += 2) {
        // ---- even sub-step: buffer 0, consume a0, prefetch a1 ----
        wait_asynccnt0();
        __syncthreads();
        // NK is even, so it+1 < NK always: stage chunk it+1 into buffer 1
        async_copy_b128(lp0 + BBUF_BYTES, gp0 + 32);
        async_copy_b128(lp1 + BBUF_BYTES, gp1 + 32);
        gp0 += 32; gp1 += 32;
        ap += 32;
        a1 = loadA(ap, hi);                 // prefetch (in-bounds tail reads)
        kstep(acc, a0, bfrag);

        // ---- odd sub-step: buffer 1, consume a1, prefetch a0 ----
        wait_asynccnt0();
        __syncthreads();
        if (it + 2 < NK) {                  // stage chunk it+2 into buffer 0
            async_copy_b128(lp0, gp0 + 32);
            async_copy_b128(lp1, gp1 + 32);
        }
        gp0 += 32; gp1 += 32;
        ap += 32;
        a0 = loadA(ap, hi);                 // prefetch (in-bounds tail reads)
        kstep(acc, a1, bfrag + BBUF_BYTES);
    }

    float gamma = __uint_as_float(*gbits);
    float wg    = (*wsum) * (1.0f / (float)W_ELEMS);
    float scale = gamma * wg * (1.0f / 128.0f);

    // C/D layout: VGPR r -> row m0 + r + hi*8, column n0 + tn*16 + lo
    int rowb = m0 + hi * 8;
    #pragma unroll
    for (int tn = 0; tn < 4; ++tn) {
        #pragma unroll
        for (int r = 0; r < 8; ++r) {
            float h = acc[tn][r] * scale;
            float g = acc[4 + tn][r] * scale;
            out[(size_t)(rowb + r) * NHALF + n0 + tn * 16 + lo] =
                h * (g / (1.0f + __expf(-g)));
        }
    }
}

// -------------------------------------------------------------------------
extern "C" void kernel_launch(void* const* d_in, const int* in_sizes, int n_in,
                              void* d_out, int out_size, void* d_ws, size_t ws_size,
                              hipStream_t stream) {
    (void)in_sizes; (void)n_in; (void)out_size; (void)ws_size;
    const float* x = (const float*)d_in[0];
    const float* w = (const float*)d_in[1];
    float* out = (float*)d_out;

    char* ws = (char*)d_ws;
    float*        wsum     = (float*)(ws + WS_WSUM);
    unsigned int* gbits    = (unsigned int*)(ws + WS_GBITS);
    float*        partials = (float*)(ws + WS_PART);
    float*        mean     = (float*)(ws + WS_MEAN);
    float*        rstd     = (float*)(ws + WS_RSTD);
    _Float16*     qa       = (_Float16*)(ws + WS_QA);
    _Float16*     wb       = (_Float16*)(ws + WS_WB);

    hipMemsetAsync(ws, 0, 16, stream);   // zero wsum + gbits accumulators

    wsum_partial_kernel<<<256, 256, 0, stream>>>(w, partials);
    wsum_final_kernel<<<1, 256, 0, stream>>>(partials, wsum);
    ln_stats_kernel<<<N_TOK, 256, 0, stream>>>(x, mean, rstd, gbits);
    quant_act_kernel<<<N_TOK, 256, 0, stream>>>(x, mean, rstd, gbits, qa);
    tern_w_kernel<<<W_ELEMS / (256 * 8), 256, 0, stream>>>(w, wsum, wb);

    dim3 grid(NHALF / 64, N_TOK / 128);  // 64 x 64 blocks
    gemm_glu_kernel<<<grid, 256, 0, stream>>>(qa, wb, wsum, gbits, out);
}
